// ENPGModel_69698729279909
// MI455X (gfx1250) — matrix operand order repaired and checked by reading, not compiled, not run on hardware
//
#include <hip/hip_runtime.h>
#include <hip/hip_bf16.h>

// ---------------------------------------------------------------------------
// Problem constants (from the reference)
// ---------------------------------------------------------------------------
#define N_NODES 10000
#define N_EDGES 160000
#define NB      64      // number of graphs (B)
#define HSZ     64      // hidden size HS
#define HA      128     // attention size
#define NK      32      // RBF kernels per filter
#define NF      64      // number of filters
#define EIND    201     // edge-MLP input features
#define EINDP   224     // padded to 7 * 32
#define NIND    163     // node-MLP input features
#define NINDP   192     // padded to 6 * 32

typedef __attribute__((ext_vector_type(16))) _Float16 v16h;
typedef __attribute__((ext_vector_type(8)))  float    v8f;
typedef __attribute__((ext_vector_type(4)))  float    vf4;
typedef __attribute__((ext_vector_type(4)))  _Float16 vh4;

// ---------------------------------------------------------------------------
// WMMA fragment helpers (wave32, 16x16x32 f16 -> f32)
// A 16x32 (MxK): lane L<16 -> row M=L, K {0..7,16..23}; lane>=16 -> K {8..15,24..31}
// B 32x16 (KxN): lane -> col N=lane&15, K {0..7,16..23} (lane<16) / {8..15,24..31}
// C/D 16x16 f32: vgpr r -> row r + ((lane>=16)?8:0), col = lane&15
// ---------------------------------------------------------------------------
__device__ __forceinline__ v16h frag_a(const _Float16* p, int ld, int lane) {
  const _Float16* r = p + (lane & 15) * ld + ((lane >> 4) << 3);
  v16h a;
#pragma unroll
  for (int j = 0; j < 8; ++j) { a[j] = r[j]; a[8 + j] = r[16 + j]; }
  return a;
}

__device__ __forceinline__ v16h frag_b(const _Float16* p, int ld, int lane) {
  const _Float16* r = p + (lane & 15) + ((lane >> 4) << 3) * ld;
  v16h b;
#pragma unroll
  for (int j = 0; j < 8; ++j) { b[j] = r[j * ld]; b[8 + j] = r[(16 + j) * ld]; }
  return b;
}

__device__ __forceinline__ v8f wmma_f16(v16h a, v16h b, v8f c) {
  return __builtin_amdgcn_wmma_f32_16x16x32_f16(false, a, false, b,
                                                (short)0, c, false, false);
}

// ---------------------------------------------------------------------------
// CDNA5 async global->LDS copy (ASYNCcnt tracked); one 16B packet per lane.
// ---------------------------------------------------------------------------
__device__ __forceinline__ unsigned lds_off(const void* p) {
  return (unsigned)(size_t)p;
}
__device__ __forceinline__ void async_cp16(unsigned lds_byte_off, const void* g) {
  asm volatile("global_load_async_to_lds_b128 %0, %1, off"
               :: "v"(lds_byte_off), "v"(g) : "memory");
}
__device__ __forceinline__ void async_wait0() {
  asm volatile("s_wait_asynccnt 0x0" ::: "memory");
}

__device__ __forceinline__ void cvt_store4(_Float16* dst, vf4 v) {
  vh4 o;
  o.x = (_Float16)v.x; o.y = (_Float16)v.y;
  o.z = (_Float16)v.z; o.w = (_Float16)v.w;
  *(vh4*)dst = o;
}

// ---------------------------------------------------------------------------
// Small utility kernels
// ---------------------------------------------------------------------------
__global__ void k_zero(float* __restrict__ p, int n) {
  int i = blockIdx.x * 256 + threadIdx.x;
  if (i < n) p[i] = 0.f;
}

__global__ void k_zero_h(_Float16* __restrict__ p, int n) {
  int i = blockIdx.x * 256 + threadIdx.x;
  if (i < n) p[i] = (_Float16)0.f;
}

__global__ void k_cvt(const float* __restrict__ src, _Float16* __restrict__ dst,
                      int n) {
  int i = blockIdx.x * 256 + threadIdx.x;
  if (i < n) dst[i] = (_Float16)src[i];
}

// pad rows [rows, prows) with zeros (K padding for WMMA chunks)
__global__ void k_cvt_pad(const float* __restrict__ src,
                          _Float16* __restrict__ dst, int rows, int cols,
                          int prows) {
  int i = blockIdx.x * 256 + threadIdx.x;
  if (i < prows * cols) {
    int r = i / cols;
    dst[i] = (r < rows) ? (_Float16)src[i] : (_Float16)0.f;
  }
}

__global__ void k_prep(const int* __restrict__ batch, int* __restrict__ goff) {
  int t = threadIdx.x;
  if (t <= NB) {
    int lo = 0, hi = N_NODES;
    while (lo < hi) {
      int mid = (lo + hi) >> 1;
      if (batch[mid] < t) lo = mid + 1; else hi = mid;
    }
    goff[t] = lo;
  }
}

// ---------------------------------------------------------------------------
// k_pos: Pmlp (3 -> 64 -> 64), scalar per node
// ---------------------------------------------------------------------------
__global__ __launch_bounds__(256) void k_pos(
    const float* __restrict__ pos,
    const float* __restrict__ w1, const float* __restrict__ b1,
    const float* __restrict__ g1, const float* __restrict__ bb1,
    const float* __restrict__ m1, const float* __restrict__ v1,
    const float* __restrict__ w2, const float* __restrict__ b2,
    float* __restrict__ outp) {
  __shared__ float sw1[3 * HSZ];
  __shared__ float ssc[HSZ], sst[HSZ], sb2[HSZ];
  __shared__ float sw2[HSZ * HSZ];
  int tid = threadIdx.x;
  for (int i = tid; i < 3 * HSZ; i += 256) sw1[i] = w1[i];
  for (int i = tid; i < HSZ * HSZ; i += 256) sw2[i] = w2[i];
  if (tid < HSZ) {
    float s = g1[tid] * rsqrtf(v1[tid] + 1e-5f);
    ssc[tid] = s;
    sst[tid] = (b1[tid] - m1[tid]) * s + bb1[tid];
    sb2[tid] = b2[tid];
  }
  __syncthreads();
  int n = blockIdx.x * 256 + tid;
  if (n >= N_NODES) return;
  float p0 = pos[n * 3 + 0], p1 = pos[n * 3 + 1], p2 = pos[n * 3 + 2];
  float h[HSZ];
#pragma unroll 8
  for (int j = 0; j < HSZ; ++j) {
    float a = p0 * sw1[j] + p1 * sw1[HSZ + j] + p2 * sw1[2 * HSZ + j];
    a = a * ssc[j] + sst[j];
    h[j] = a > 0.f ? a : 0.f;
  }
  for (int o = 0; o < HSZ; ++o) {
    float acc = sb2[o];
#pragma unroll 8
    for (int j = 0; j < HSZ; ++j) acc += h[j] * sw2[j * HSZ + o];
    outp[n * HSZ + o] = acc;
  }
}

// ---------------------------------------------------------------------------
// k_edge: edge MLP.  64 edges/block, 4 waves.  Weights pre-converted f16;
// K-chunks of W1 DMA'd into double-buffered LDS (async) overlapped with WMMA.
// ---------------------------------------------------------------------------
__global__ __launch_bounds__(128) void k_edge(
    const float* __restrict__ x, const int* __restrict__ ei,
    const float* __restrict__ ea, const float* __restrict__ pe,
    const float* __restrict__ u, const int* __restrict__ batch,
    const _Float16* __restrict__ w1h,   // [EINDP x HSZ] f16, zero-padded rows
    const _Float16* __restrict__ w2h,   // [HSZ x HSZ] f16
    const float* __restrict__ b1,
    const float* __restrict__ g1, const float* __restrict__ bb1,
    const float* __restrict__ m1, const float* __restrict__ v1,
    const float* __restrict__ b2,
    float* __restrict__ e_out, float* __restrict__ agg_e) {
  __shared__ __attribute__((aligned(16))) _Float16 s_in[64 * EINDP];  // 28 KB
  __shared__ __attribute__((aligned(16))) _Float16 s_w[2][32 * HSZ];  //  8 KB
  __shared__ __attribute__((aligned(16))) _Float16 s_h[64 * HSZ];     //  8 KB
  __shared__ __attribute__((aligned(16))) _Float16 s_w2[HSZ * HSZ];   //  8 KB
  const int tid = threadIdx.x;
  const int lane = tid & 31;
  const int w = tid >> 5;
  const int e0 = blockIdx.x * 64;

  {  // async DMA: full W2 (8KB) and W1 chunk 0 (4KB)
    const char* g2 = (const char*)w2h;
    unsigned l2 = lds_off(s_w2);
#pragma unroll
    for (int it = 0; it < 4; ++it)
      async_cp16(l2 + (unsigned)(it * 128 + tid) * 16u, g2 + (it * 128 + tid) * 16);
    const char* g1c = (const char*)w1h;
    unsigned l1 = lds_off(&s_w[0][0]);
    async_cp16(l1 + (unsigned)tid * 16u, g1c + tid * 16);
    async_cp16(l1 + 2048u + (unsigned)tid * 16u, g1c + 2048 + tid * 16);
  }

  {  // gather e_in tile (64 x 224) as f16, 2 threads per edge
    const int el = tid >> 1, half = tid & 1;
    const int e = e0 + el;
    int r = 0, c = 0;
    if (e < N_EDGES) { r = ei[e]; c = ei[N_EDGES + e]; }
    const float* xr = x + r * 64;
    const float* xc = x + c * 64;
    _Float16* din = s_in + el * EINDP;
    vf4 z; z.x = 0.f; z.y = 0.f; z.z = 0.f; z.w = 0.f;
#pragma unroll
    for (int q = 0; q < 20; ++q) {  // f in [0,160): x|x|edge_attr
      int f = (half * 20 + q) * 4;
      vf4 v = z;
      if (e < N_EDGES) {
        if (f < 64)       v = *(const vf4*)(xr + f);
        else if (f < 128) v = *(const vf4*)(xc + (f - 64));
        else              v = *(const vf4*)(ea + e * 32 + (f - 128));
      }
      cvt_store4(din + f, v);
    }
    int bu = (e < N_EDGES) ? batch[r] : 0;
#pragma unroll
    for (int j = 0; j < 32; ++j) {  // f in [160,224): pe parts, u, pad
      int f = 160 + half * 32 + j;
      float val = 0.f;
      if (e < N_EDGES) {
        if (f < 163)       val = pe[r * 3 + (f - 160)];
        else if (f < 166)  val = pe[c * 3 + (f - 163)];
        else if (f < 169)  val = pe[c * 3 + (f - 166)] - pe[r * 3 + (f - 166)];
        else if (f < EIND) val = u[bu * 32 + (f - 169)];
      }
      din[f] = (_Float16)val;
    }
  }

  // GEMM1: [64 x 224] @ [224 x 64], pipelined W1 chunks
  v8f c1[4] = {};
  for (int kc = 0; kc < 7; ++kc) {
    async_wait0();
    __syncthreads();
    if (kc + 1 < 7) {
      const char* g = (const char*)(w1h + (kc + 1) * 32 * HSZ);
      unsigned l = lds_off(&s_w[(kc + 1) & 1][0]);
      async_cp16(l + (unsigned)tid * 16u, g + tid * 16);
      async_cp16(l + 2048u + (unsigned)tid * 16u, g + 2048 + tid * 16);
    }
    v16h a = frag_a(&s_in[(w * 16) * EINDP + kc * 32], EINDP, lane);
#pragma unroll
    for (int t = 0; t < 4; ++t) {
      v16h b = frag_b(&s_w[kc & 1][t * 16], HSZ, lane);
      c1[t] = wmma_f16(a, b, c1[t]);
    }
  }
  {  // BN1 (folded) + relu -> f16 hidden (wave-private rows)
    int rb = (lane >> 4) << 3;
#pragma unroll
    for (int t = 0; t < 4; ++t) {
      int col = t * 16 + (lane & 15);
      float s = g1[col] * rsqrtf(v1[col] + 1e-5f);
      float sh = (b1[col] - m1[col]) * s + bb1[col];
#pragma unroll
      for (int r = 0; r < 8; ++r) {
        float h = c1[t][r] * s + sh;
        s_h[(w * 16 + rb + r) * HSZ + col] = (_Float16)(h > 0.f ? h : 0.f);
      }
    }
  }
  v8f c2[4] = {};
#pragma unroll
  for (int kc = 0; kc < 2; ++kc) {
    v16h a = frag_a(&s_h[(w * 16) * HSZ + kc * 32], HSZ, lane);
#pragma unroll
    for (int t = 0; t < 4; ++t) {
      v16h b = frag_b(&s_w2[(kc * 32) * HSZ + t * 16], HSZ, lane);
      c2[t] = wmma_f16(a, b, c2[t]);
    }
  }
  {  // +b2, write e_out, scatter-add agg_e[dst]
    int rb = (lane >> 4) << 3;
#pragma unroll
    for (int t = 0; t < 4; ++t) {
      int col = t * 16 + (lane & 15);
      float bias = b2[col];
#pragma unroll
      for (int r = 0; r < 8; ++r) {
        int e = e0 + w * 16 + rb + r;
        if (e < N_EDGES) {
          float val = c2[t][r] + bias;
          e_out[e * HSZ + col] = val;
          int dst = ei[N_EDGES + e];
          atomicAdd(&agg_e[dst * HSZ + col], val);
        }
      }
    }
  }
}

// ---------------------------------------------------------------------------
// k_node: node MLP.  n_in = [x | posemb | u[batch] | agg_e] (163 -> 192 pad)
// Also writes an f16 mirror of n_out for the RBFA DMA path.
// ---------------------------------------------------------------------------
__global__ __launch_bounds__(128) void k_node(
    const float* __restrict__ x, const float* __restrict__ pe,
    const float* __restrict__ u, const int* __restrict__ batch,
    const float* __restrict__ agg_e,
    const _Float16* __restrict__ w1h,   // [NINDP x HSZ] f16 zero-padded
    const _Float16* __restrict__ w2h,   // [HSZ x HSZ] f16
    const float* __restrict__ b1,
    const float* __restrict__ g1, const float* __restrict__ bb1,
    const float* __restrict__ m1, const float* __restrict__ v1,
    const float* __restrict__ b2,
    const float* __restrict__ g2, const float* __restrict__ bb2,
    const float* __restrict__ m2, const float* __restrict__ v2,
    float* __restrict__ n_out, _Float16* __restrict__ n_outh,
    float* __restrict__ pooled) {
  __shared__ __attribute__((aligned(16))) _Float16 s_in[64 * NINDP];  // 24 KB
  __shared__ __attribute__((aligned(16))) _Float16 s_w[2][32 * HSZ];
  __shared__ __attribute__((aligned(16))) _Float16 s_h[64 * HSZ];
  __shared__ __attribute__((aligned(16))) _Float16 s_w2[HSZ * HSZ];
  const int tid = threadIdx.x;
  const int lane = tid & 31;
  const int w = tid >> 5;
  const int n0 = blockIdx.x * 64;

  {
    const char* gg = (const char*)w2h;
    unsigned l2 = lds_off(s_w2);
#pragma unroll
    for (int it = 0; it < 4; ++it)
      async_cp16(l2 + (unsigned)(it * 128 + tid) * 16u, gg + (it * 128 + tid) * 16);
    const char* g1c = (const char*)w1h;
    unsigned l1 = lds_off(&s_w[0][0]);
    async_cp16(l1 + (unsigned)tid * 16u, g1c + tid * 16);
    async_cp16(l1 + 2048u + (unsigned)tid * 16u, g1c + 2048 + tid * 16);
  }

  {  // gather n_in, 2 threads per node
    const int nl = tid >> 1, half = tid & 1;
    const int n = n0 + nl;
    _Float16* din = s_in + nl * NINDP;
    vf4 z; z.x = 0.f; z.y = 0.f; z.z = 0.f; z.w = 0.f;
#pragma unroll
    for (int q = 0; q < 8; ++q) {  // f in [0,64): x
      int f = (half * 8 + q) * 4;
      vf4 v = z;
      if (n < N_NODES) v = *(const vf4*)(x + n * 64 + f);
      cvt_store4(din + f, v);
    }
    int bu = (n < N_NODES) ? batch[n] : 0;
#pragma unroll
    for (int j = 0; j < 64; ++j) {  // f in [64,192)
      int f = 64 + half * 64 + j;
      float val = 0.f;
      if (n < N_NODES) {
        if (f < 67)        val = pe[n * 3 + (f - 64)];
        else if (f < 99)   val = u[bu * 32 + (f - 67)];
        else if (f < NIND) val = agg_e[n * HSZ + (f - 99)];
      }
      din[f] = (_Float16)val;
    }
  }

  v8f c1[4] = {};
  for (int kc = 0; kc < 6; ++kc) {
    async_wait0();
    __syncthreads();
    if (kc + 1 < 6) {
      const char* g = (const char*)(w1h + (kc + 1) * 32 * HSZ);
      unsigned l = lds_off(&s_w[(kc + 1) & 1][0]);
      async_cp16(l + (unsigned)tid * 16u, g + tid * 16);
      async_cp16(l + 2048u + (unsigned)tid * 16u, g + 2048 + tid * 16);
    }
    v16h a = frag_a(&s_in[(w * 16) * NINDP + kc * 32], NINDP, lane);
#pragma unroll
    for (int t = 0; t < 4; ++t) {
      v16h b = frag_b(&s_w[kc & 1][t * 16], HSZ, lane);
      c1[t] = wmma_f16(a, b, c1[t]);
    }
  }
  {
    int rb = (lane >> 4) << 3;
#pragma unroll
    for (int t = 0; t < 4; ++t) {
      int col = t * 16 + (lane & 15);
      float s = g1[col] * rsqrtf(v1[col] + 1e-5f);
      float sh = (b1[col] - m1[col]) * s + bb1[col];
#pragma unroll
      for (int r = 0; r < 8; ++r) {
        float h = c1[t][r] * s + sh;
        s_h[(w * 16 + rb + r) * HSZ + col] = (_Float16)(h > 0.f ? h : 0.f);
      }
    }
  }
  v8f c2[4] = {};
#pragma unroll
  for (int kc = 0; kc < 2; ++kc) {
    v16h a = frag_a(&s_h[(w * 16) * HSZ + kc * 32], HSZ, lane);
#pragma unroll
    for (int t = 0; t < 4; ++t) {
      v16h b = frag_b(&s_w2[(kc * 32) * HSZ + t * 16], HSZ, lane);
      c2[t] = wmma_f16(a, b, c2[t]);
    }
  }
  {  // +b2 then BN2 (folded); write n_out (f32+f16) and pooled_n scatter
    int rb = (lane >> 4) << 3;
#pragma unroll
    for (int t = 0; t < 4; ++t) {
      int col = t * 16 + (lane & 15);
      float s = g2[col] * rsqrtf(v2[col] + 1e-5f);
      float sh = (b2[col] - m2[col]) * s + bb2[col];
#pragma unroll
      for (int r = 0; r < 8; ++r) {
        int n = n0 + w * 16 + rb + r;
        if (n < N_NODES) {
          float val = c2[t][r] * s + sh;
          n_out[n * HSZ + col] = val;
          n_outh[n * HSZ + col] = (_Float16)val;
          atomicAdd(&pooled[batch[n] * HSZ + col], val);
        }
      }
    }
  }
}

// ---------------------------------------------------------------------------
// k_rbfa: one block per (graph b, filter f); feat[32x128] in WMMA registers:
//   x1   = n_out_chunk[32x64] @ x1w[64x128]      (WMMA)
//   feat += rbf_chunk^T[32x32] @ x1[32x128]      (WMMA)
// x1w[f] and the f16 n_out chunks arrive via double-buffered async DMA.
// ---------------------------------------------------------------------------
__global__ __launch_bounds__(128) void k_rbfa(
    const float* __restrict__ pos, const int* __restrict__ goff,
    const _Float16* __restrict__ n_outh,  // [(N+32) x HSZ] f16, zero-padded
    const _Float16* __restrict__ x1wh,    // [NF x HSZ x HA] f16
    const float* __restrict__ centers, const float* __restrict__ widths,
    const float* __restrict__ x1b, const float* __restrict__ aw,
    const float* __restrict__ ab, const float* __restrict__ x2w,
    const float* __restrict__ x2b, const float* __restrict__ lng,
    const float* __restrict__ lnb, const float* __restrict__ x3w,
    const float* __restrict__ x3b, float* __restrict__ fouts) {
  const int b = blockIdx.x;
  const int f = blockIdx.y;
  __shared__ __attribute__((aligned(16))) _Float16 s_x1w[HSZ * HA];    // 16 KB
  __shared__ __attribute__((aligned(16))) _Float16 s_nh[2][32 * HSZ];  //  8 KB
  __shared__ __attribute__((aligned(16))) _Float16 s_rbf[NK * 32];     //  2 KB
  __shared__ __attribute__((aligned(16))) _Float16 s_x1[32 * HA];      //  8 KB
  __shared__ float s_feat[NK * HA];                                    // 16 KB
  __shared__ float s_cen[NK * 3];
  __shared__ float s_wd[NK];
  __shared__ float s_sc[NK];
  __shared__ float s_agg[HA];
  __shared__ float s_hv[HSZ];
  const int tid = threadIdx.x, lane = tid & 31, w = tid >> 5;
  const int rt = w & 1;   // row tile (node rows / k rows 0..15 or 16..31)
  const int cg = w >> 1;  // column group (64 of the 128 HA columns)

  {  // async DMA x1w[f] -> LDS (16 KB)
    const char* g = (const char*)(x1wh + f * HSZ * HA);
    unsigned l = lds_off(s_x1w);
#pragma unroll
    for (int it = 0; it < 8; ++it)
      async_cp16(l + (unsigned)(it * 128 + tid) * 16u, g + (it * 128 + tid) * 16);
  }
  for (int i = tid; i < NK * 3; i += 128) s_cen[i] = centers[f * NK * 3 + i];
  if (tid < NK) s_wd[tid] = widths[f * NK + tid];

  const int s0 = goff[b], s1 = goff[b + 1];
  if (s0 < s1) {  // async DMA first n_out chunk (4 KB)
    const char* g = (const char*)(n_outh + s0 * HSZ);
    unsigned l = lds_off(&s_nh[0][0]);
    async_cp16(l + (unsigned)tid * 16u, g + tid * 16);
    async_cp16(l + 2048u + (unsigned)tid * 16u, g + 2048 + tid * 16);
  }

  v8f cf[4] = {};
  int it = 0;
  for (int nb = s0; nb < s1; nb += 32, ++it) {
    const int cur = it & 1;
    async_wait0();
    __syncthreads();
    if (nb + 32 < s1) {  // DMA next node chunk into the other buffer
      const char* g = (const char*)(n_outh + (nb + 32) * HSZ);
      unsigned l = lds_off(&s_nh[cur ^ 1][0]);
      async_cp16(l + (unsigned)tid * 16u, g + tid * 16);
      async_cp16(l + 2048u + (unsigned)tid * 16u, g + 2048 + tid * 16);
    }
    {  // rbf^T [k][node]: thread owns node (tid&31), computes 8 k's
      const int nl = tid & 31;
      const int n = nb + nl;
      const bool valid = n < s1;
      float px = 0.f, py = 0.f, pz = 0.f;
      if (valid) {
        px = pos[n * 3 + 0]; py = pos[n * 3 + 1]; pz = pos[n * 3 + 2];
      }
#pragma unroll
      for (int kk = 0; kk < 8; ++kk) {
        int k = (tid >> 5) * 8 + kk;
        float rv = 0.f;
        if (valid) {
          float dx = px - s_cen[k * 3 + 0];
          float dy = py - s_cen[k * 3 + 1];
          float dz = pz - s_cen[k * 3 + 2];
          float d = sqrtf(dx * dx + dy * dy + dz * dz);
          rv = __expf(-d / (s_wd[k] * s_wd[k]));
        }
        s_rbf[k * 32 + nl] = (_Float16)rv;
      }
    }
    __syncthreads();
    v8f cx[4] = {};
#pragma unroll
    for (int kc = 0; kc < 2; ++kc) {
      v16h a = frag_a(&s_nh[cur][(rt * 16) * HSZ + kc * 32], HSZ, lane);
#pragma unroll
      for (int j = 0; j < 4; ++j) {
        v16h bb = frag_b(&s_x1w[(kc * 32) * HA + (cg * 4 + j) * 16], HA, lane);
        cx[j] = wmma_f16(a, bb, cx[j]);
      }
    }
    {  // + x1b -> f16 chunk (per-wave 16x64 tile)
      int rb = (lane >> 4) << 3;
#pragma unroll
      for (int j = 0; j < 4; ++j) {
        int col = (cg * 4 + j) * 16 + (lane & 15);
        float bias = x1b[f * HA + col];
#pragma unroll
        for (int r = 0; r < 8; ++r)
          s_x1[(rt * 16 + rb + r) * HA + col] = (_Float16)(cx[j][r] + bias);
      }
    }
    __syncthreads();
    {  // feat += rbf^T @ x1
      v16h a = frag_a(&s_rbf[(rt * 16) * 32], 32, lane);
#pragma unroll
      for (int j = 0; j < 4; ++j) {
        v16h bb = frag_b(&s_x1[(cg * 4 + j) * 16], HA, lane);
        cf[j] = wmma_f16(a, bb, cf[j]);
      }
    }
  }
  async_wait0();  // retire x1w DMA even if the graph was empty
  __syncthreads();
  {  // dump feat accumulators
    int rb = (lane >> 4) << 3;
#pragma unroll
    for (int j = 0; j < 4; ++j) {
      int col = (cg * 4 + j) * 16 + (lane & 15);
#pragma unroll
      for (int r = 0; r < 8; ++r)
        s_feat[(rt * 16 + rb + r) * HA + col] = cf[j][r];
    }
  }
  __syncthreads();
  if (tid < NK) {  // attention logits
    float acc = ab[f];
    for (int a = 0; a < HA; ++a) acc += s_feat[tid * HA + a] * aw[f * HA + a];
    s_sc[tid] = acc;
  }
  __syncthreads();
  if (tid < HA) {  // softmax over NK, weighted sum
    float mx = -1e30f;
    for (int k = 0; k < NK; ++k) mx = fmaxf(mx, s_sc[k]);
    float den = 0.f;
    for (int k = 0; k < NK; ++k) den += __expf(s_sc[k] - mx);
    float acc = 0.f;
    for (int k = 0; k < NK; ++k)
      acc += (__expf(s_sc[k] - mx) / den) * s_feat[k * HA + tid];
    s_agg[tid] = acc;
  }
  __syncthreads();
  if (tid < HSZ) {  // x2 + leaky
    float acc = x2b[f * HSZ + tid];
    for (int a = 0; a < HA; ++a)
      acc += s_agg[a] * x2w[f * HA * HSZ + a * HSZ + tid];
    s_hv[tid] = acc > 0.f ? acc : 0.2f * acc;
  }
  __syncthreads();
  if (tid == 0) {  // LayerNorm + x3
    float mu = 0.f;
    for (int j = 0; j < HSZ; ++j) mu += s_hv[j];
    mu /= HSZ;
    float va = 0.f;
    for (int j = 0; j < HSZ; ++j) { float d = s_hv[j] - mu; va += d * d; }
    va /= HSZ;
    float inv = rsqrtf(va + 1e-5f);
    float acc = x3b[f];
    for (int j = 0; j < HSZ; ++j) {
      float hn = (s_hv[j] - mu) * inv * lng[f * HSZ + j] + lnb[f * HSZ + j];
      acc += hn * x3w[f * HSZ + j];
    }
    fouts[b * NF + f] = acc;
  }
}

// ---------------------------------------------------------------------------
// k_final: rbfa_mlp (leaky, bn1+bn2) then Gmlp (relu). One thread per graph.
// ---------------------------------------------------------------------------
__global__ __launch_bounds__(64) void k_final(
    const float* __restrict__ u, const float* __restrict__ fouts,
    const float* __restrict__ pooled,
    const float* __restrict__ Rw1, const float* __restrict__ Rb1,
    const float* __restrict__ Rg1, const float* __restrict__ Rbb1,
    const float* __restrict__ Rm1, const float* __restrict__ Rv1,
    const float* __restrict__ Rw2, const float* __restrict__ Rb2,
    const float* __restrict__ Rg2, const float* __restrict__ Rbb2,
    const float* __restrict__ Rm2, const float* __restrict__ Rv2,
    const float* __restrict__ Gw1, const float* __restrict__ Gb1,
    const float* __restrict__ Gg1, const float* __restrict__ Gbb1,
    const float* __restrict__ Gm1, const float* __restrict__ Gv1,
    const float* __restrict__ Gw2, const float* __restrict__ Gb2,
    float* __restrict__ u_out) {
  int b = threadIdx.x;
  float h1[64], h2[64], g1v[64];
  for (int j = 0; j < 64; ++j) {
    float acc = Rb1[j];
    for (int i = 0; i < 64; ++i) acc += fouts[b * 64 + i] * Rw1[i * 64 + j];
    float s = Rg1[j] * rsqrtf(Rv1[j] + 1e-5f);
    acc = (acc - Rm1[j]) * s + Rbb1[j];
    h1[j] = acc > 0.f ? acc : 0.2f * acc;  // leaky
  }
  for (int o = 0; o < 64; ++o) {
    float acc = Rb2[o];
    for (int j = 0; j < 64; ++j) acc += h1[j] * Rw2[j * 64 + o];
    float s = Rg2[o] * rsqrtf(Rv2[o] + 1e-5f);
    h2[o] = (acc - Rm2[o]) * s + Rbb2[o];  // u_rbfa
  }
  for (int j = 0; j < 64; ++j) {
    float acc = Gb1[j];
    for (int i = 0; i < 32; ++i) acc += u[b * 32 + i] * Gw1[i * 64 + j];
    for (int i = 0; i < 64; ++i) acc += h2[i] * Gw1[(32 + i) * 64 + j];
    for (int i = 0; i < 64; ++i) acc += pooled[b * 64 + i] * Gw1[(96 + i) * 64 + j];
    float s = Gg1[j] * rsqrtf(Gv1[j] + 1e-5f);
    acc = (acc - Gm1[j]) * s + Gbb1[j];
    g1v[j] = acc > 0.f ? acc : 0.f;
  }
  for (int o = 0; o < 64; ++o) {
    float acc = Gb2[o];
    for (int j = 0; j < 64; ++j) acc += g1v[j] * Gw2[j * 64 + o];
    u_out[b * 64 + o] = acc;
  }
}

// ---------------------------------------------------------------------------
// Host-side launch.  d_in order = DFS insertion order of setup_inputs():
//  0 x 1 edge_index 2 edge_attr 3 pos 4 posemb 5 u 6 batch
//  Emlp: 7 w1 8 b1 9..12 bn1{g,b,m,v} 13 w2 14 b2
//  Nmlp: 15 w1 16 b1 17..20 bn1 21 w2 22 b2 23..26 bn2
//  Gmlp: 27 w1 28 b1 29..32 bn1 33 w2 34 b2
//  Pmlp: 35 w1 36 b1 37..40 bn1 41 w2 42 b2
//  rbfa_mlp: 43 w1 44 b1 45..48 bn1 49 w2 50 b2 51..54 bn2
//  rbfa: 55 centers 56 widths 57 x1w 58 x1b 59 aw 60 ab 61 x2w 62 x2b
//        63 lng 64 lnb 65 x3w 66 x3b
// Output (concat): n_out | e_out | u_out | posemb_new
// ---------------------------------------------------------------------------
extern "C" void kernel_launch(void* const* d_in, const int* in_sizes, int n_in,
                              void* d_out, int out_size, void* d_ws,
                              size_t ws_size, hipStream_t stream) {
  (void)in_sizes; (void)n_in; (void)out_size; (void)ws_size;
#define FP(i) ((const float*)d_in[(i)])
#define IP(i) ((const int*)d_in[(i)])

  const float* x = FP(0);
  const int* ei = IP(1);
  const float* ea = FP(2);
  const float* pos = FP(3);
  const float* pe = FP(4);
  const float* u = FP(5);
  const int* batch = IP(6);

  // workspace layout
  char* wsb = (char*)d_ws;
  float* agg_e = (float*)wsb;                  // N*HS f32
  float* pooled = agg_e + N_NODES * HSZ;       // B*HS
  float* fouts = pooled + NB * HSZ;            // B*NF
  int* goff = (int*)(fouts + NB * NF);         // B+1
  size_t off =
      (((size_t)(N_NODES * HSZ + NB * HSZ + NB * NF) * 4 + (NB + 1) * 4) + 511) &
      ~(size_t)511;
  _Float16* Ew1h = (_Float16*)(wsb + off);     // EINDP*HSZ
  _Float16* Nw1h = Ew1h + EINDP * HSZ;         // NINDP*HSZ
  _Float16* Ew2h = Nw1h + NINDP * HSZ;         // HSZ*HSZ
  _Float16* Nw2h = Ew2h + HSZ * HSZ;           // HSZ*HSZ
  _Float16* x1wh = Nw2h + HSZ * HSZ;           // NF*HSZ*HA
  _Float16* n_outh = x1wh + NF * HSZ * HA;     // (N+32)*HSZ f16

  // outputs
  float* o_n = (float*)d_out;
  float* o_e = o_n + N_NODES * HSZ;
  float* o_u = o_e + N_EDGES * HSZ;
  float* o_p = o_u + NB * HSZ;

  int zn = N_NODES * HSZ + NB * HSZ + NB * NF;
  k_zero<<<(zn + 255) / 256, 256, 0, stream>>>((float*)d_ws, zn);
  // zero the 32-row f16 tail pad of n_outh (chunk overreads land here)
  k_zero_h<<<(32 * HSZ + 255) / 256, 256, 0, stream>>>(n_outh + N_NODES * HSZ,
                                                       32 * HSZ);
  k_prep<<<1, 128, 0, stream>>>(batch, goff);

  // f16 weight packing (once per launch; deterministic)
  k_cvt_pad<<<(EINDP * HSZ + 255) / 256, 256, 0, stream>>>(FP(7), Ew1h, EIND,
                                                           HSZ, EINDP);
  k_cvt_pad<<<(NINDP * HSZ + 255) / 256, 256, 0, stream>>>(FP(15), Nw1h, NIND,
                                                           HSZ, NINDP);
  k_cvt<<<(HSZ * HSZ + 255) / 256, 256, 0, stream>>>(FP(13), Ew2h, HSZ * HSZ);
  k_cvt<<<(HSZ * HSZ + 255) / 256, 256, 0, stream>>>(FP(21), Nw2h, HSZ * HSZ);
  k_cvt<<<(NF * HSZ * HA + 255) / 256, 256, 0, stream>>>(FP(57), x1wh,
                                                         NF * HSZ * HA);

  k_pos<<<(N_NODES + 255) / 256, 256, 0, stream>>>(
      pos, FP(35), FP(36), FP(37), FP(38), FP(39), FP(40), FP(41), FP(42), o_p);

  k_edge<<<(N_EDGES + 63) / 64, 128, 0, stream>>>(
      x, ei, ea, pe, u, batch, Ew1h, Ew2h,
      FP(8), FP(9), FP(10), FP(11), FP(12), FP(14), o_e, agg_e);

  k_node<<<(N_NODES + 63) / 64, 128, 0, stream>>>(
      x, pe, u, batch, agg_e, Nw1h, Nw2h,
      FP(16), FP(17), FP(18), FP(19), FP(20),
      FP(22), FP(23), FP(24), FP(25), FP(26), o_n, n_outh, pooled);

  dim3 gr(NB, NF);
  k_rbfa<<<gr, 128, 0, stream>>>(
      pos, goff, n_outh, x1wh,
      FP(55), FP(56), FP(58), FP(59), FP(60),
      FP(61), FP(62), FP(63), FP(64), FP(65), FP(66), fouts);

  k_final<<<1, 64, 0, stream>>>(
      u, fouts, pooled,
      FP(43), FP(44), FP(45), FP(46), FP(47), FP(48),
      FP(49), FP(50), FP(51), FP(52), FP(53), FP(54),
      FP(27), FP(28), FP(29), FP(30), FP(31), FP(32), FP(33), FP(34),
      o_u);
#undef FP
#undef IP
}
